// Encoder_26517128085785
// MI455X (gfx1250) — matrix operand order repaired and checked by reading, not compiled, not run on hardware
//
#include <hip/hip_runtime.h>
#include <hip/hip_bf16.h>
#include <math.h>

// Problem constants
#define B_   64
#define T_   50
#define N_   24
#define F_   256
#define H_   256
#define G4H  1024
#define KC   512   // concatenated K (=F+H) for the per-step gate GEMM

typedef __bf16 bf16_t;
typedef __attribute__((ext_vector_type(8)))  __bf16 v8bf;
typedef __attribute__((ext_vector_type(16))) __bf16 v16bf;
typedef __attribute__((ext_vector_type(8)))  float  v8f;

union ABu { v16bf v; v8bf h[2]; };

__device__ __forceinline__ float sigmoidf_(float x) { return 1.0f / (1.0f + __expf(-x)); }
__device__ __forceinline__ float tanh_fast(float x) { return 1.0f - 2.0f / (__expf(2.0f * x) + 1.0f); }

// ---------------------------------------------------------------------------
// Kernel 1: convert weights to bf16; stack [W_ih | W_hh] row-wise into a
// single [1024][512] matrix; fuse the two gate biases.
// ---------------------------------------------------------------------------
__global__ void prep_weights_kernel(const float* __restrict__ W_ih,
                                    const float* __restrict__ W_hh,
                                    const float* __restrict__ b_ih,
                                    const float* __restrict__ b_hh,
                                    const float* __restrict__ W_h1,
                                    const float* __restrict__ W_h2,
                                    const float* __restrict__ W_fc,
                                    bf16_t* __restrict__ Wcat,
                                    bf16_t* __restrict__ Wh1,
                                    bf16_t* __restrict__ Wh2,
                                    bf16_t* __restrict__ Wfc,
                                    float*  __restrict__ b4)
{
    int idx = blockIdx.x * 256 + threadIdx.x;
    if (idx < G4H * KC) {
        int n = idx >> 9, k = idx & 511;
        float v = (k < F_) ? W_ih[n * F_ + k] : W_hh[n * H_ + (k - F_)];
        Wcat[idx] = (bf16_t)v;
        return;
    }
    int r = idx - G4H * KC;
    if (r < H_ * F_) { Wh1[r] = (bf16_t)W_h1[r]; return; }
    r -= H_ * F_;
    if (r < H_ * F_) { Wh2[r] = (bf16_t)W_h2[r]; return; }
    r -= H_ * F_;
    if (r < H_ * H_) { Wfc[r] = (bf16_t)W_fc[r]; return; }
    r -= H_ * H_;
    if (r < G4H) { b4[r] = b_ih[r] + b_hh[r]; }
}

// ---------------------------------------------------------------------------
// Kernel 2: persistent graph-LSTM. One workgroup (8 wave32) per batch.
//   h state: LDS (24x256 f32).  c state: accumulator registers (v8f tiles).
//   Per step: A = [G*x_t | G*h] staged as bf16 32x512 in LDS (rows 24..31
//   zero-padded), gates = A @ [W_ih|W_hh]^T + b via v_wmma_f32_16x16x32_bf16.
//   Each of the wave's four 16x16 output tiles is produced by its own K-loop
//   holding just acc[4] (32 VGPRs) -> low pressure, no spills, no vgpr-msb.
// ---------------------------------------------------------------------------
__global__ __launch_bounds__(256, 1) void encoder_persist_kernel(
    const float*  __restrict__ x,      // [B][T][24][256]
    const float*  __restrict__ G,      // [24][24]
    const float*  __restrict__ b_h1,
    const float*  __restrict__ b_h2,
    const float*  __restrict__ b_fc,
    const bf16_t* __restrict__ Wcat,   // [1024][512]
    const bf16_t* __restrict__ Wh1,    // [256][256]
    const bf16_t* __restrict__ Wh2,    // [256][256]
    const bf16_t* __restrict__ Wfc,    // [256][256]
    const float*  __restrict__ b4,     // [1024] = b_ih + b_hh
    float*        __restrict__ out)    // [B][24][256]
{
    __shared__ float sG[N_ * N_];
    __shared__ float sh[N_ * H_];
    __shared__ __align__(16) bf16_t sA[32 * KC];

    const int tid  = threadIdx.x;
    const int lane = tid & 31;
    const int w    = tid >> 5;       // wave id 0..7
    const int b    = blockIdx.x;     // batch
    const int lo16 = lane & 15;
    const int hi   = lane >> 4;

    for (int i = tid; i < N_ * N_; i += 256) sG[i] = G[i];
    for (int i = tid; i < 32 * KC; i += 256) sA[i] = (bf16_t)0.0f;  // zero-pad rows 24..31
    __syncthreads();

    // ---- WMMA operand loaders (contiguous k order in A and B: k = hi*16 + s) ----
    auto loadA = [&](int mt, int kt) -> v16bf {   // 16x32 bf16 A tile from LDS
        const int row = mt * 16 + lo16;
        const bf16_t* p = &sA[row * KC + kt * 32 + hi * 16];
        ABu a;
        a.h[0] = *(const v8bf*)(p);        // K = hi*16 .. +7
        a.h[1] = *(const v8bf*)(p + 8);    // K = hi*16+8 .. +15
        return a.v;
    };
    auto loadB = [&](const bf16_t* W, int ncol0, int stride, int kt) -> v16bf { // 32x16 B tile
        const bf16_t* p = W + (size_t)(ncol0 + lo16) * stride + kt * 32 + hi * 16;
        ABu bm;
        bm.h[0] = *(const v8bf*)(p);       // K = hi*16 .. +7
        bm.h[1] = *(const v8bf*)(p + 8);   // K = hi*16+8 .. +15
        return bm.v;
    };

    // ---- node mixes: Xg = G * x_t, Hg = G * h (K=24, done in VALU) ----
    auto mixX = [&](int t) {  // -> sA cols [0,256)
        const float* xp = x + (((size_t)b * T_ + t) * N_) * F_ + tid;
        float xv[N_];
        #pragma unroll
        for (int m = 0; m < N_; ++m) xv[m] = xp[m * F_];
        #pragma unroll
        for (int n = 0; n < N_; ++n) {
            float a = 0.0f;
            #pragma unroll
            for (int m = 0; m < N_; ++m) a += sG[n * N_ + m] * xv[m];
            sA[n * KC + tid] = (bf16_t)a;
        }
    };
    auto mixH = [&](int off) {  // -> sA cols [off, off+256)
        float hv[N_];
        #pragma unroll
        for (int m = 0; m < N_; ++m) hv[m] = sh[m * H_ + tid];
        #pragma unroll
        for (int n = 0; n < N_; ++n) {
            float a = 0.0f;
            #pragma unroll
            for (int m = 0; m < N_; ++m) a += sG[n * N_ + m] * hv[m];
            sA[n * KC + off + tid] = (bf16_t)a;
        }
    };

    // ---- N=256, K=256 GEMM (init h0/c0 and final projection) ----
    auto gemm256 = [&](const bf16_t* W, const float* bias, v8f (&acc)[2][2]) {
        #pragma unroll
        for (int jt = 0; jt < 2; ++jt) {
            v8f a2[2];
            float bv = bias[w * 32 + jt * 16 + lo16];
            #pragma unroll
            for (int mt = 0; mt < 2; ++mt)
                #pragma unroll
                for (int r = 0; r < 8; ++r) a2[mt][r] = bv;

            #pragma unroll 2
            for (int kt = 0; kt < 8; ++kt) {
                v16bf a0 = loadA(0, kt);
                v16bf a1 = loadA(1, kt);
                v16bf bt = loadB(W, w * 32 + jt * 16, H_, kt);
                a2[0] = __builtin_amdgcn_wmma_f32_16x16x32_bf16(
                    false, a0, false, bt, (short)0, a2[0], false, false);
                a2[1] = __builtin_amdgcn_wmma_f32_16x16x32_bf16(
                    false, a1, false, bt, (short)0, a2[1], false, false);
            }
            acc[0][jt] = a2[0];
            acc[1][jt] = a2[1];
        }
    };

    // ---- init: h0 = (G x0) W_h1^T + b_h1 ;  c0 = (G x0) W_h2^T + b_h2 ----
    mixX(0);
    __syncthreads();

    v8f hcc[2][2];
    gemm256(Wh1, b_h1, hcc);
    #pragma unroll
    for (int mt = 0; mt < 2; ++mt)
        #pragma unroll
        for (int jt = 0; jt < 2; ++jt) {
            int colj = w * 32 + jt * 16 + lo16;
            #pragma unroll
            for (int r = 0; r < 8; ++r) {
                int m = mt * 16 + hi * 8 + r;
                if (m < N_) sh[m * H_ + colj] = hcc[mt][jt][r];
            }
        }

    v8f creg[2][2];
    gemm256(Wh2, b_h2, creg);
    __syncthreads();

    // ---- sequential recurrence ----
    #pragma unroll 1
    for (int t = 0; t < T_; ++t) {
        mixX(t);     // A cols [0,256)   = G * x_t  (bf16)
        mixH(F_);    // A cols [256,512) = G * h    (bf16)
        __syncthreads();

        if (t + 1 < T_)  // warm L2/WGP$ for next x tile
            __builtin_prefetch(x + (((size_t)b * T_ + (t + 1)) * N_) * F_ + tid, 0, 0);

        // gates = A @ Wcat^T + b4.  Wave w owns hidden cols [w*32, w*32+32)
        // of all four gate blocks.  One K-loop per (jt, mt) output tile with
        // only acc[4] live; jt/mt fully unrolled so creg indices are constant.
        #pragma unroll
        for (int jt = 0; jt < 2; ++jt) {
            #pragma unroll
            for (int mt = 0; mt < 2; ++mt) {
                v8f acc[4];
                #pragma unroll
                for (int gb = 0; gb < 4; ++gb) {
                    float bv = b4[gb * H_ + w * 32 + jt * 16 + lo16];
                    #pragma unroll
                    for (int r = 0; r < 8; ++r) acc[gb][r] = bv;
                }

                #pragma unroll 2
                for (int kt = 0; kt < KC / 32; ++kt) {
                    v16bf a = loadA(mt, kt);
                    #pragma unroll
                    for (int gb = 0; gb < 4; ++gb) {
                        v16bf bt = loadB(Wcat, gb * H_ + w * 32 + jt * 16, KC, kt);
                        acc[gb] = __builtin_amdgcn_wmma_f32_16x16x32_bf16(
                            false, a, false, bt, (short)0, acc[gb], false, false);
                    }
                }

                // LSTM pointwise for this 16x16 tile, all in registers
                int colj = w * 32 + jt * 16 + lo16;
                #pragma unroll
                for (int r = 0; r < 8; ++r) {
                    float iv = acc[0][r];
                    float fv = acc[1][r];
                    float gv = acc[2][r];
                    float ov = acc[3][r];
                    float cn = sigmoidf_(fv) * creg[mt][jt][r] + sigmoidf_(iv) * tanh_fast(gv);
                    creg[mt][jt][r] = cn;
                    float hv = sigmoidf_(ov) * tanh_fast(cn);
                    int m = mt * 16 + hi * 8 + r;
                    if (m < N_) sh[m * H_ + colj] = hv;
                }
            }
        }
        __syncthreads();
    }

    // ---- output: tanh((G h) W_fc^T + b_fc) ----
    mixH(0);
    __syncthreads();
    v8f fo[2][2];
    gemm256(Wfc, b_fc, fo);
    float* outp = out + (size_t)b * N_ * H_;
    #pragma unroll
    for (int mt = 0; mt < 2; ++mt)
        #pragma unroll
        for (int jt = 0; jt < 2; ++jt) {
            int colj = w * 32 + jt * 16 + lo16;
            #pragma unroll
            for (int r = 0; r < 8; ++r) {
                int m = mt * 16 + hi * 8 + r;
                if (m < N_) outp[m * H_ + colj] = tanh_fast(fo[mt][jt][r]);
            }
        }
}

// ---------------------------------------------------------------------------
extern "C" void kernel_launch(void* const* d_in, const int* in_sizes, int n_in,
                              void* d_out, int out_size, void* d_ws, size_t ws_size,
                              hipStream_t stream)
{
    (void)in_sizes; (void)n_in; (void)out_size; (void)ws_size;

    const float* x    = (const float*)d_in[0];
    const float* G    = (const float*)d_in[1];
    const float* W_ih = (const float*)d_in[2];
    const float* b_ih = (const float*)d_in[3];
    const float* W_hh = (const float*)d_in[4];
    const float* b_hh = (const float*)d_in[5];
    const float* W_h1 = (const float*)d_in[6];
    const float* b_h1 = (const float*)d_in[7];
    const float* W_h2 = (const float*)d_in[8];
    const float* b_h2 = (const float*)d_in[9];
    const float* W_fc = (const float*)d_in[10];
    const float* b_fc = (const float*)d_in[11];

    char* ws = (char*)d_ws;
    bf16_t* Wcat = (bf16_t*)ws; ws += (size_t)G4H * KC * sizeof(bf16_t);  // 1 MB
    bf16_t* Wh1  = (bf16_t*)ws; ws += (size_t)H_ * F_ * sizeof(bf16_t);   // 128 KB
    bf16_t* Wh2  = (bf16_t*)ws; ws += (size_t)H_ * F_ * sizeof(bf16_t);
    bf16_t* Wfc  = (bf16_t*)ws; ws += (size_t)H_ * H_ * sizeof(bf16_t);
    float*  b4   = (float*)ws;                                            // 4 KB

    int total = G4H * KC + 3 * H_ * F_ + G4H;
    prep_weights_kernel<<<(total + 255) / 256, 256, 0, stream>>>(
        W_ih, W_hh, b_ih, b_hh, W_h1, W_h2, W_fc, Wcat, Wh1, Wh2, Wfc, b4);

    encoder_persist_kernel<<<B_, 256, 0, stream>>>(
        x, G, b_h1, b_h2, b_fc, Wcat, Wh1, Wh2, Wfc, b4, (float*)d_out);
}
